// NeuroNet_38362647888397
// MI455X (gfx1250) — matrix operand order repaired and checked by reading, not compiled, not run on hardware
//
#include <hip/hip_runtime.h>
#include <hip/hip_bf16.h>
#include <math.h>

// Problem constants (from reference)
#define B_   8
#define P_   256
#define F_   64
#define F2_  128
#define T_   32640           // P*(P-1)/2
#define NC_  10
#define HW_  128
#define PS_  8

#define PAIRS_PER_BLK 128
#define BLKS_PER_B    (T_ / PAIRS_PER_BLK)   // 255

typedef __attribute__((ext_vector_type(16))) _Float16 v16h;
typedef __attribute__((ext_vector_type(8)))  float    v8f;

__device__ __forceinline__ float gelu_f(float x) {
    // jax.nn.gelu default (approximate=True, tanh form)
    float x3 = x * x * x;
    return 0.5f * x * (1.0f + tanhf(0.7978845608028654f * (x + 0.044715f * x3)));
}

// Recover (row, col) of strict-lower-triangle pair index t (np.tril_indices order):
// t = r*(r-1)/2 + c, c < r
__device__ __forceinline__ void pair_rc(int t, int& r, int& c) {
    int rr = (int)((1.0f + sqrtf(8.0f * (float)t + 1.0f)) * 0.5f);
    while (rr > 1 && rr * (rr - 1) / 2 > t) --rr;
    while ((rr + 1) * rr / 2 <= t) ++rr;
    r = rr;
    c = t - rr * (rr - 1) / 2;
}

// ---------------- Stage 0: patch conv (a 64-dot per output) + gelu + pos_emb ----------------
__global__ void k_patch(const float* __restrict__ x, const float* __restrict__ cw,
                        const float* __restrict__ cb, const float* __restrict__ pos,
                        float* __restrict__ hT) {
    __shared__ float patch[64];
    int blk = blockIdx.x;            // b*P + p
    int b = blk >> 8, p = blk & 255;
    int py = p >> 4, px = p & 15;
    int t = threadIdx.x;             // 0..63 (also the output feature f)
    int ky = t >> 3, kx = t & 7;
    patch[t] = x[(b * HW_ + py * PS_ + ky) * HW_ + px * PS_ + kx];
    __syncthreads();
    float acc = cb[t];
#pragma unroll
    for (int k = 0; k < 64; ++k) acc += patch[k] * cw[t * 64 + k];
    hT[(b * P_ + p) * F_ + t] = gelu_f(acc) + pos[t * P_ + p];
}

// ---------------- Stage 1: e = 2 * LN(gelu(hT @ emb_w + emb_b)) ----------------
__global__ void k_embed(const float* __restrict__ hT, const float* __restrict__ ew,
                        const float* __restrict__ eb, const float* __restrict__ g1,
                        const float* __restrict__ b1, float* __restrict__ e) {
    __shared__ float hrow[64];
    __shared__ float gbuf[64];
    int bp = blockIdx.x;
    int f = threadIdx.x;             // 0..63
    hrow[f] = hT[bp * F_ + f];
    __syncthreads();
    float acc = eb[f];
#pragma unroll
    for (int k = 0; k < 64; ++k) acc += hrow[k] * ew[k * F_ + f];
    float g = gelu_f(acc);
    gbuf[f] = g;
    __syncthreads();
    float s = 0.f, q = 0.f;
    for (int k = 0; k < 64; ++k) { float v = gbuf[k]; s += v; q += v * v; }
    float mu  = s * (1.0f / 64.0f);
    float var = q * (1.0f / 64.0f) - mu * mu;
    float rs  = rsqrtf(var + 1e-5f);
    e[bp * F_ + f] = ((g - mu) * rs * g1[f] + b1[f]) * 2.0f;
}

// -------- Stage 2: fold MLP layer-1 + inter_w through the concat: A=e@W1_top, Bm=e@W1_bot --------
__global__ void k_ab(const float* __restrict__ e, const float* __restrict__ w1,
                     const float* __restrict__ iw, float* __restrict__ A,
                     float* __restrict__ Bm, float* __restrict__ sr, float* __restrict__ sc) {
    __shared__ float erow[64];
    int bp = blockIdx.x;
    int n = threadIdx.x;             // 0..127
    if (n < 64) erow[n] = e[bp * F_ + n];
    __syncthreads();
    float a = 0.f, bmv = 0.f;
#pragma unroll 8
    for (int k = 0; k < 64; ++k) {
        float ek = erow[k];
        a   += ek * w1[k * F2_ + n];
        bmv += ek * w1[(64 + k) * F2_ + n];
    }
    A[bp * F2_ + n]  = a;
    Bm[bp * F2_ + n] = bmv;
    if (n == 0) { float s = 0.f; for (int k = 0; k < 64; ++k) s += erow[k] * iw[k];      sr[bp] = s; }
    if (n == 1) { float s = 0.f; for (int k = 0; k < 64; ++k) s += erow[k] * iw[64 + k]; sc[bp] = s; }
}

// ---- W2 -> f16 (row-major [K=128][N=128]) and epilogue scalars C1 = g3.iw, C2 = b3.iw ----
__global__ void k_w2h(const float* __restrict__ w2, const float* __restrict__ g3,
                      const float* __restrict__ b3, const float* __restrict__ iw,
                      _Float16* __restrict__ w2h, float* __restrict__ cst) {
    int i = blockIdx.x * blockDim.x + threadIdx.x;
    if (i < F2_ * F2_) w2h[i] = (_Float16)w2[i];
    if (i == 0) {
        float c1 = 0.f, c2 = 0.f;
        for (int n = 0; n < F2_; ++n) { c1 += g3[n] * iw[n]; c2 += b3[n] * iw[n]; }
        cst[0] = c1; cst[1] = c2;
    }
}

// ---------------- Stage 3 (the heavy one): per-pair gelu -> WMMA GEMM @W2 -> fused LN+score ----------------
// Block = 256 threads (8 waves). Block owns 128 pairs; wave w owns output columns [16w, 16w+16).
// score = rs*(Sum(m*g3*iw) - mu*C1) + C2 + sr[row] + sc[col]   (LN algebraically folded)
__global__ void k_mlp(const float* __restrict__ A, const float* __restrict__ Bm,
                      const float* __restrict__ b1, const _Float16* __restrict__ w2h,
                      const float* __restrict__ b2, const float* __restrict__ g3,
                      const float* __restrict__ iw, const float* __restrict__ cst,
                      const float* __restrict__ sr, const float* __restrict__ sc,
                      float* __restrict__ score) {
    __shared__ _Float16 ush[PAIRS_PER_BLK * F2_];   // u in f16, row-major [pair][k]  (32 KB)
    __shared__ int   ri[PAIRS_PER_BLK];
    __shared__ int   ci[PAIRS_PER_BLK];
    __shared__ float pbias[PAIRS_PER_BLK];
    __shared__ float redS[16][8];
    __shared__ float redQ[16][8];
    __shared__ float redG[16][8];

    int b     = blockIdx.x / BLKS_PER_B;
    int tile0 = (blockIdx.x % BLKS_PER_B) * PAIRS_PER_BLK;
    int tid   = threadIdx.x;

    // Warm L2/L0 for the W2 fragment gathers (global_prefetch_b8)
    __builtin_prefetch((const void*)(w2h + (tid << 6)), 0, 1);

    if (tid < PAIRS_PER_BLK) {
        int t = tile0 + tid, r, c;
        pair_rc(t, r, c);
        ri[tid] = r; ci[tid] = c;
        pbias[tid] = sr[b * P_ + r] + sc[b * P_ + c];
    }
    __syncthreads();

    // u[pair][k] = gelu(A[row] + Bm[col] + b1)  (layer-1 folded through the concat)
    for (int idx = tid; idx < PAIRS_PER_BLK * F2_; idx += 256) {
        int pr = idx >> 7, k = idx & 127;
        float v = A[(b * P_ + ri[pr]) * F2_ + k] + Bm[(b * P_ + ci[pr]) * F2_ + k] + b1[k];
        ush[idx] = (_Float16)gelu_f(v);
    }
    __syncthreads();

    int w    = tid >> 5;          // wave -> N tile
    int lane = tid & 31;
    int hi   = lane >> 4;
    int col  = w * 16 + (lane & 15);
    int mr   = lane & 15;

    // B fragments of W2 for this wave's columns. ISA 16-bit B layout (32x16 tile):
// lane = N column; half j <-> K = kc*32 + hi*8 + (j&7) + 16*(j>>3)
    v16h bf[4];
#pragma unroll
    for (int kc = 0; kc < 4; ++kc)
#pragma unroll
        for (int j = 0; j < 16; ++j)
            bf[kc][j] = w2h[(kc * 32 + hi * 8 + (j & 7) + ((j >> 3) << 4)) * F2_ + col];

    float b2c = b2[col];
    float gw  = g3[col] * iw[col];          // per-column weight for the fused score dot
    float c1  = cst[0], c2 = cst[1];

    for (int mt = 0; mt < PAIRS_PER_BLK / 16; ++mt) {
        int pb = mt * 16;
        const _Float16* ub = &ush[(pb + mr) * F2_];

        v8f acc = {};
#pragma unroll
        for (int kc = 0; kc < 4; ++kc) {
            // ISA 16-bit A layout (16x32 tile): lane m; halves 0..7 <-> K = kc*32+hi*8+{0..7},
            // halves 8..15 <-> K = kc*32+hi*8+{16..23}
            v16h af;
            int kb = kc * 32 + hi * 8;
#pragma unroll
            for (int j = 0; j < 8; ++j) {
                af[j]     = ub[kb + j];
                af[8 + j] = ub[kb + 16 + j];
            }
            acc = __builtin_amdgcn_wmma_f32_16x16x32_f16(false, af, false, bf[kc],
                                                         (short)0, acc, false, false);
        }

        // C/D layout: acc[g] is element (row = g + 8*hi, col).
        // Single fused reduction phase: S=sum(m), Q=sum(m^2), G=sum(m*g3*iw).
        float ps[8], pq[8], pg[8];
#pragma unroll
        for (int g = 0; g < 8; ++g) {
            float m = acc[g] + b2c;
            ps[g] = m;
            pq[g] = m * m;
            pg[g] = m * gw;
        }
        // reduce across the 16 lanes of this half (xor masks < 16 never cross halves)
#pragma unroll
        for (int off = 1; off < 16; off <<= 1)
#pragma unroll
            for (int g = 0; g < 8; ++g) {
                ps[g] += __shfl_xor(ps[g], off);
                pq[g] += __shfl_xor(pq[g], off);
                pg[g] += __shfl_xor(pg[g], off);
            }
        if ((lane & 15) == 0)
#pragma unroll
            for (int g = 0; g < 8; ++g) {
                redS[hi * 8 + g][w] = ps[g];
                redQ[hi * 8 + g][w] = pq[g];
                redG[hi * 8 + g][w] = pg[g];
            }
        __syncthreads();
        if (tid < 16) {
            float s1 = 0.f, s2 = 0.f, sg = 0.f;
#pragma unroll
            for (int k = 0; k < 8; ++k) { s1 += redS[tid][k]; s2 += redQ[tid][k]; sg += redG[tid][k]; }
            float mu  = s1 * (1.0f / 128.0f);
            float var = s2 * (1.0f / 128.0f) - mu * mu;
            float rs  = rsqrtf(var + 1e-5f);
            // score = LN(m)@iw + (pc@iw decomposed as sr[row]+sc[col])
            score[b * T_ + tile0 + pb + tid] = rs * (sg - mu * c1) + c2 + pbias[pb + tid];
        }
        __syncthreads();   // protect red* reuse next mt
    }
}

// ---------------- Stage 4: per-batch softmax stats over T ----------------
__global__ void k_softmax_stats(const float* __restrict__ score, float* __restrict__ stats) {
    __shared__ float red[256];
    int b = blockIdx.x, tid = threadIdx.x;
    const float* s = score + b * T_;
    float mx = -1e30f;
    for (int t = tid; t < T_; t += 256) mx = fmaxf(mx, s[t]);
    red[tid] = mx; __syncthreads();
    for (int o = 128; o > 0; o >>= 1) { if (tid < o) red[tid] = fmaxf(red[tid], red[tid + o]); __syncthreads(); }
    mx = red[0]; __syncthreads();
    float sm = 0.f;
    for (int t = tid; t < T_; t += 256) sm += expf(s[t] - mx);
    red[tid] = sm; __syncthreads();
    for (int o = 128; o > 0; o >>= 1) { if (tid < o) red[tid] += red[tid + o]; __syncthreads(); }
    if (tid == 0) { stats[b * 2] = mx; stats[b * 2 + 1] = red[0]; }
}

// ---------------- Stage 5: write attn output ----------------
__global__ void k_attn(const float* __restrict__ score, const float* __restrict__ stats,
                       float* __restrict__ attn) {
    int i = blockIdx.x * blockDim.x + threadIdx.x;   // 0..B*T-1 (exact)
    int b = i / T_;
    attn[i] = expf(score[i] - stats[b * 2]) / stats[b * 2 + 1];
}

// ---------------- Stage 6: deterministic per-patch attention mass (pooled decomposition) ----------------
// wr[b][r] = sum_{c<r} attn[t(r,c)] (contiguous run); wc[b][c] = sum_{r>c} attn[t(r,c)]
__global__ void k_wsum(const float* __restrict__ attn, float* __restrict__ wr, float* __restrict__ wc) {
    int b = blockIdx.x, i = threadIdx.x;   // i = patch index 0..255
    const float* a = attn + b * T_;
    float s = 0.f;
    for (int c = 0; c < i; ++c) s += a[i * (i - 1) / 2 + c];
    wr[b * P_ + i] = s;
    float s2 = 0.f;
    for (int r = i + 1; r < P_; ++r) s2 += a[r * (r - 1) / 2 + i];
    wc[b * P_ + i] = s2;
}

// ---------------- Stage 7: pooled = [e^T wr | e^T wc]; y = pooled @ pred_w + pred_b ----------------
__global__ void k_final(const float* __restrict__ e, const float* __restrict__ wr,
                        const float* __restrict__ wc, const float* __restrict__ pw,
                        const float* __restrict__ pbv, float* __restrict__ y) {
    __shared__ float pl[128];
    int b = blockIdx.x, f = threadIdx.x;   // 0..127
    const float* eb_ = e + b * P_ * F_;
    float s = 0.f;
    if (f < 64) {
        for (int r = 0; r < P_; ++r) s += wr[b * P_ + r] * eb_[r * F_ + f];
    } else {
        int ff = f - 64;
        for (int c = 0; c < P_; ++c) s += wc[b * P_ + c] * eb_[c * F_ + ff];
    }
    pl[f] = s;
    __syncthreads();
    if (f < NC_) {
        float acc = pbv[f];
        for (int k = 0; k < F2_; ++k) acc += pl[k] * pw[k * NC_ + f];
        y[b * NC_ + f] = acc;
    }
}

extern "C" void kernel_launch(void* const* d_in, const int* in_sizes, int n_in,
                              void* d_out, int out_size, void* d_ws, size_t ws_size,
                              hipStream_t stream) {
    const float* x      = (const float*)d_in[0];
    const float* conv_w = (const float*)d_in[1];
    const float* conv_b = (const float*)d_in[2];
    const float* pos    = (const float*)d_in[3];
    const float* emb_w  = (const float*)d_in[4];
    const float* emb_b  = (const float*)d_in[5];
    const float* ln1_g  = (const float*)d_in[6];
    const float* ln1_b  = (const float*)d_in[7];
    const float* w1     = (const float*)d_in[8];
    const float* mb1    = (const float*)d_in[9];
    const float* w2     = (const float*)d_in[10];
    const float* mb2    = (const float*)d_in[11];
    const float* g3     = (const float*)d_in[12];
    const float* b3     = (const float*)d_in[13];
    const float* iw     = (const float*)d_in[14];
    const float* pw     = (const float*)d_in[15];
    const float* pbv    = (const float*)d_in[16];

    float* ws  = (float*)d_ws;
    float* hT  = ws;                  // 131072
    float* e   = hT + 131072;         // 131072
    float* A   = e  + 131072;         // 262144
    float* Bm  = A  + 262144;         // 262144
    float* sr  = Bm + 262144;         // 2048
    float* sc  = sr + 2048;           // 2048
    float* scr = sc + 2048;           // 261120 (scores)
    float* wr  = scr + 261120;        // 2048
    float* wc  = wr + 2048;           // 2048
    float* st  = wc + 2048;           // 16 (softmax stats)
    float* cst = st + 16;             // 16 (C1, C2)
    _Float16* w2h = (_Float16*)(cst + 16);  // 16384 halfs

    float* y    = (float*)d_out;      // [B, NC] first
    float* attn = y + B_ * NC_;       // then [B, T]

    k_patch        <<<B_ * P_,           64,  0, stream>>>(x, conv_w, conv_b, pos, hT);
    k_embed        <<<B_ * P_,           64,  0, stream>>>(hT, emb_w, emb_b, ln1_g, ln1_b, e);
    k_ab           <<<B_ * P_,           128, 0, stream>>>(e, w1, iw, A, Bm, sr, sc);
    k_w2h          <<<64,                256, 0, stream>>>(w2, g3, b3, iw, w2h, cst);
    k_mlp          <<<B_ * BLKS_PER_B,   256, 0, stream>>>(A, Bm, mb1, w2h, mb2, g3, iw, cst, sr, sc, scr);
    k_softmax_stats<<<B_,                256, 0, stream>>>(scr, st);
    k_attn         <<<(B_ * T_) / 256,   256, 0, stream>>>(scr, st, attn);
    k_wsum         <<<B_,                256, 0, stream>>>(attn, wr, wc);
    k_final        <<<B_,                128, 0, stream>>>(e, wr, wc, pw, pbv, y);
}
